// CrossAttentionFusion_68281390072581
// MI455X (gfx1250) — matrix operand order repaired and checked by reading, not compiled, not run on hardware
//
#include <hip/hip_runtime.h>
#include <hip/hip_fp16.h>

#define B_   4
#define N_   2048
#define C1_  256
#define C2_  384
#define OUT_ 512
#define H_   8
#define D_   64
#define M_   (B_ * N_)   // 8192 rows

typedef __attribute__((ext_vector_type(16))) _Float16 v16h;
typedef __attribute__((ext_vector_type(8)))  float    v8f;

union Frag { v16h v; unsigned int u[8]; };

__device__ __forceinline__ v8f wmma_f16(v16h a, v16h b, v8f c) {
  // D = A(16x32 f16) x B(32x16 f16) + C(16x16 f32)
  return __builtin_amdgcn_wmma_f32_16x16x32_f16(false, a, false, b, (short)0, c, false, false);
}

// A-fragment k order used by this code (contiguous): lanes0-15 K=0..15, lanes16-31 K=16..31, 2 K per VGPR
__device__ __forceinline__ int fragKA(int j, int g) {
  return (g << 4) + (j << 1);
}
// B-fragment k order used by this code (contiguous, same as A): lanes0-15 K=0..15, lanes16-31 K=16..31, 2 K per VGPR
__device__ __forceinline__ int fragKB(int j, int g) { return (g << 4) + (j << 1); }

// ---------------------------------------------------------------------------
// elementwise helpers
// ---------------------------------------------------------------------------
__global__ void cvt_f2h(const float* __restrict__ src, __half* __restrict__ dst, int n) {
  int i = blockIdx.x * 256 + threadIdx.x;
  if (i < n) dst[i] = __float2half(src[i]);
}

// Pack W[K,Nc] (f32, row-major) into fragment-major f16 pair dwords:
//   Wp[((k>>5)*Nc + col)*16 + q],  q = g*8+j,  holds (W[k32+16g+2j][col], W[..+1][col])
// so each lane's 8 B-fragment dwords are contiguous (2x global_load_b128).
__global__ void pack_w(const float* __restrict__ W, unsigned int* __restrict__ Wp,
                       int K, int Nc) {
  int i = blockIdx.x * 256 + threadIdx.x;
  if (i >= (K >> 1) * Nc) return;
  int q   = i & 15;
  int rem = i >> 4;           // kb*Nc + col
  int col = rem % Nc;
  int kb  = rem / Nc;
  int k   = kb * 32 + ((q >> 3) << 4) + ((q & 7) << 1);
  __half lo = __float2half(W[(size_t)k * Nc + col]);
  __half hi = __float2half(W[(size_t)(k + 1) * Nc + col]);
  Wp[i] = (unsigned int)__half_as_ushort(lo) | ((unsigned int)__half_as_ushort(hi) << 16);
}

// ---------------------------------------------------------------------------
// GEMM: out[M,Nc] = Ah[M,K](f16) * W[K,Nc] + bias ; per-wave 16x64 strip
// mode bits: 1 = write f32, 2 = write f16 (row-major),
//            4 = sigmoid epilogue, 8 = write f16 pair-packed (for attention V)
// grid: (Nc/64, M/64), block: 128 (4 waves)
// ---------------------------------------------------------------------------
__global__ void gemm_wmma(const __half* __restrict__ Ah, const unsigned int* __restrict__ Wp,
                          const float* __restrict__ bias, float* __restrict__ outF,
                          __half* __restrict__ outH, int K, int Nc, int mode) {
  const int lane = threadIdx.x & 31;
  const int wave = threadIdx.x >> 5;
  const int n = lane & 15, g = lane >> 4;
  const int m0 = blockIdx.y * 64 + wave * 16;
  const int n0 = blockIdx.x * 64;
  const int mrow = m0 + n;  // A-fragment row for this lane

  v8f c0 = {0.f,0.f,0.f,0.f,0.f,0.f,0.f,0.f};
  v8f c1 = c0, c2 = c0, c3 = c0;

  const unsigned int* Arow = (const unsigned int*)(Ah + (size_t)mrow * K);
  for (int ks = 0; ks < K; ks += 32) {
    Frag a;
#pragma unroll
    for (int j = 0; j < 8; ++j) a.u[j] = Arow[(ks + fragKA(j, g)) >> 1];

    // fragment-major weights: lane reads 8 contiguous dwords per tile
    const unsigned int* Wbase =
        Wp + ((size_t)(ks >> 5) * Nc + n0 + n) * 16 + g * 8;
    Frag b;
#pragma unroll
    for (int j = 0; j < 8; ++j) b.u[j] = Wbase[j];
    c0 = wmma_f16(a.v, b.v, c0);
#pragma unroll
    for (int j = 0; j < 8; ++j) b.u[j] = Wbase[256 + j];      // +16 cols
    c1 = wmma_f16(a.v, b.v, c1);
#pragma unroll
    for (int j = 0; j < 8; ++j) b.u[j] = Wbase[512 + j];      // +32 cols
    c2 = wmma_f16(a.v, b.v, c2);
#pragma unroll
    for (int j = 0; j < 8; ++j) b.u[j] = Wbase[768 + j];      // +48 cols
    c3 = wmma_f16(a.v, b.v, c3);
  }

  v8f cs[4] = {c0, c1, c2, c3};
#pragma unroll
  for (int t = 0; t < 4; ++t) {
#pragma unroll
    for (int r = 0; r < 8; ++r) {
      int row = m0 + r + 8 * g;      // C/D layout: M = r + 8*(lane>=16)
      int col = n0 + 16 * t + n;
      float val = cs[t][r] + bias[col];
      if (mode & 4) val = __fdividef(1.f, 1.f + __expf(-val));
      if (mode & 1) outF[(size_t)row * Nc + col] = val;
      if (mode & 2) outH[(size_t)row * Nc + col] = __float2half(val);
      if (mode & 8) {
        // pair-packed: dword (row/2)*Nc+col holds rows (2k, 2k+1) as (lo, hi)
        size_t pidx = ((size_t)(row >> 1) * Nc + col) * 2 + (row & 1);
        ((unsigned short*)outH)[pidx] = __half_as_ushort(__float2half(val));
      }
    }
  }
}

// ---------------------------------------------------------------------------
// Flash attention: softmax(Q K^T / 8) V, head_dim 64, N=2048 keys.
// grid: (N/64, H, B), block: 128 (4 waves; one 16-query tile per wave)
// Q/K are f16 [B*N, OUT] row-major; V is pair-packed dwords; Out f32.
// ---------------------------------------------------------------------------
__global__ void attn_wmma(const __half* __restrict__ Qh, const __half* __restrict__ Kh,
                          const __half* __restrict__ Vpk, float* __restrict__ Out) {
  const int lane = threadIdx.x & 31;
  const int wave = threadIdx.x >> 5;
  const int n = lane & 15, g = lane >> 4;
  const int b = blockIdx.z, h = blockIdx.y;
  const int q0 = (blockIdx.x * 4 + wave) * 16;
  const int hc = h * D_;

  __shared__ float pbuf[4][16][33];  // per-wave P tile, padded stride

  // Q A-fragments for head-dim 0..31 and 32..63
  const unsigned int* Qrow =
      (const unsigned int*)(Qh + (size_t)(b * N_ + q0 + n) * OUT_ + hc);
  Frag qa0, qa1;
#pragma unroll
  for (int j = 0; j < 8; ++j) {
    qa0.u[j] = Qrow[fragKA(j, g) >> 1];
    qa1.u[j] = Qrow[(32 + fragKA(j, g)) >> 1];
  }

  v8f acc0 = {0.f,0.f,0.f,0.f,0.f,0.f,0.f,0.f};
  v8f acc1 = acc0, acc2 = acc0, acc3 = acc0;
  float Mx[8], L[8];
#pragma unroll
  for (int r = 0; r < 8; ++r) { Mx[r] = -1e30f; L[r] = 0.f; }

  const unsigned int* Vp = (const unsigned int*)Vpk;

  for (int k0 = 0; k0 < N_; k0 += 32) {
    // ---- scores: two 16-key tiles, K-dim = 64 (2 wmma each) ----
    v8f s0 = {0.f,0.f,0.f,0.f,0.f,0.f,0.f,0.f};
    v8f s1 = s0;
    const unsigned int* Krow0 =
        (const unsigned int*)(Kh + (size_t)(b * N_ + k0 + n) * OUT_ + hc);
    const unsigned int* Krow1 =
        (const unsigned int*)(Kh + (size_t)(b * N_ + k0 + 16 + n) * OUT_ + hc);
    Frag kb;
#pragma unroll
    for (int j = 0; j < 8; ++j) kb.u[j] = Krow0[8 * g + j];
    s0 = wmma_f16(qa0.v, kb.v, s0);
#pragma unroll
    for (int j = 0; j < 8; ++j) kb.u[j] = Krow0[16 + 8 * g + j];
    s0 = wmma_f16(qa1.v, kb.v, s0);
#pragma unroll
    for (int j = 0; j < 8; ++j) kb.u[j] = Krow1[8 * g + j];
    s1 = wmma_f16(qa0.v, kb.v, s1);
#pragma unroll
    for (int j = 0; j < 8; ++j) kb.u[j] = Krow1[16 + 8 * g + j];
    s1 = wmma_f16(qa1.v, kb.v, s1);

    // ---- online softmax (row = r + 8*g lives across 16 lanes of one half) ----
#pragma unroll
    for (int r = 0; r < 8; ++r) {
      float a = s0[r] * 0.125f, bb = s1[r] * 0.125f;  // 1/sqrt(64)
      float mb = fmaxf(a, bb);
#pragma unroll
      for (int msk = 1; msk <= 8; msk <<= 1) mb = fmaxf(mb, __shfl_xor(mb, msk, 32));
      float Mn = fmaxf(Mx[r], mb);
      float alpha = __expf(Mx[r] - Mn);
      Mx[r] = Mn;
      float p0 = __expf(a - Mn), p1 = __expf(bb - Mn);
      float rs = p0 + p1;
#pragma unroll
      for (int msk = 1; msk <= 8; msk <<= 1) rs += __shfl_xor(rs, msk, 32);
      L[r] = L[r] * alpha + rs;
      acc0[r] *= alpha; acc1[r] *= alpha; acc2[r] *= alpha; acc3[r] *= alpha;
      pbuf[wave][r + 8 * g][n]      = p0;
      pbuf[wave][r + 8 * g][16 + n] = p1;
    }
    __syncthreads();

    // ---- re-shape P (C-layout) into an A-fragment via LDS ----
    Frag pf;
#pragma unroll
    for (int j = 0; j < 8; ++j) {
      int kk = fragKA(j, g);
      __half lo = __float2half(pbuf[wave][n][kk]);
      __half hi = __float2half(pbuf[wave][n][kk + 1]);
      pf.u[j] = (unsigned int)__half_as_ushort(lo) |
                ((unsigned int)__half_as_ushort(hi) << 16);
    }
    __syncthreads();

    // ---- P(16x32) @ V(32x64) in four 16-col chunks (pair-packed V) ----
#pragma unroll
    for (int t = 0; t < 4; ++t) {
      Frag vb;
#pragma unroll
      for (int j = 0; j < 8; ++j) {
        // dword = (keys k0+16g+2j, +1) at column hc+16t+n
        vb.u[j] = Vp[(size_t)(b * (N_ / 2) + (k0 >> 1) + 8 * g + j) * OUT_ +
                     hc + 16 * t + n];
      }
      if      (t == 0) acc0 = wmma_f16(pf.v, vb.v, acc0);
      else if (t == 1) acc1 = wmma_f16(pf.v, vb.v, acc1);
      else if (t == 2) acc2 = wmma_f16(pf.v, vb.v, acc2);
      else             acc3 = wmma_f16(pf.v, vb.v, acc3);
    }
  }

#pragma unroll
  for (int r = 0; r < 8; ++r) {
    int row = q0 + r + 8 * g;
    float inv = __fdividef(1.f, L[r]);
    size_t base = (size_t)(b * N_ + row) * OUT_ + hc + n;
    Out[base]      = acc0[r] * inv;
    Out[base + 16] = acc1[r] * inv;
    Out[base + 32] = acc2[r] * inv;
    Out[base + 48] = acc3[r] * inv;
  }
}

// ---------------------------------------------------------------------------
// gate input: concat[out1, out2, pn2+pt] -> f16 [M, 1536]
// ---------------------------------------------------------------------------
__global__ void build_cat3(const float* __restrict__ o1, const float* __restrict__ o2,
                           const float* __restrict__ pn2, const float* __restrict__ pt,
                           __half* __restrict__ dst) {
  int i = blockIdx.x * 256 + threadIdx.x;
  if (i >= M_ * 1536) return;
  int row = i / 1536, c = i - row * 1536;
  float v;
  if (c < 512)       v = o1[(size_t)row * 512 + c];
  else if (c < 1024) v = o2[(size_t)row * 512 + (c - 512)];
  else { int cc = c - 1024; v = pn2[(size_t)row * 512 + cc] + pt[(size_t)row * 512 + cc]; }
  dst[i] = __float2half(v);
}

// fused: concat[out1*gate, out2*(1-gate)] -> f16 [M, 1024]
__global__ void build_cat2(const float* __restrict__ o1, const float* __restrict__ o2,
                           const float* __restrict__ gate, __half* __restrict__ dst) {
  int i = blockIdx.x * 256 + threadIdx.x;
  if (i >= M_ * 1024) return;
  int row = i / 1024, c = i - row * 1024;
  float v;
  if (c < 512) { float gv = gate[(size_t)row * 512 + c]; v = o1[(size_t)row * 512 + c] * gv; }
  else { int cc = c - 512; float gv = gate[(size_t)row * 512 + cc];
         v = o2[(size_t)row * 512 + cc] * (1.f - gv); }
  dst[i] = __float2half(v);
}

// LN(x)*g+b then exact-erf GELU -> f16 ; one block per row of 512
__global__ void ln_gelu(const float* __restrict__ x, const float* __restrict__ gamma,
                        const float* __restrict__ beta, __half* __restrict__ outh) {
  int row = blockIdx.x, t = threadIdx.x;
  __shared__ float s1[256], s2[256];
  const float* xr = x + (size_t)row * 512;
  float a = xr[t], bb = xr[t + 256];
  s1[t] = a + bb; s2[t] = a * a + bb * bb;
  __syncthreads();
  for (int st = 128; st > 0; st >>= 1) {
    if (t < st) { s1[t] += s1[t + st]; s2[t] += s2[t + st]; }
    __syncthreads();
  }
  float mean = s1[0] * (1.f / 512.f);
  float var  = s2[0] * (1.f / 512.f) - mean * mean;
  float rstd = rsqrtf(var + 1e-5f);
  __half* orow = outh + (size_t)row * 512;
  float y0 = (a  - mean) * rstd * gamma[t]       + beta[t];
  float y1 = (bb - mean) * rstd * gamma[t + 256] + beta[t + 256];
  y0 = 0.5f * y0 * (1.f + erff(y0 * 0.70710678118654752f));
  y1 = 0.5f * y1 * (1.f + erff(y1 * 0.70710678118654752f));
  orow[t]       = __float2half(y0);
  orow[t + 256] = __float2half(y1);
}

// final: LN(h2 + pn2 + pt)*g + b -> f32 output ; one block per row
__global__ void final_ln(const float* __restrict__ h2, const float* __restrict__ pn2,
                         const float* __restrict__ pt, const float* __restrict__ gamma,
                         const float* __restrict__ beta, float* __restrict__ out) {
  int row = blockIdx.x, t = threadIdx.x;
  __shared__ float s1[256], s2[256];
  size_t base = (size_t)row * 512;
  float a  = h2[base + t]       + pn2[base + t]       + pt[base + t];
  float bb = h2[base + t + 256] + pn2[base + t + 256] + pt[base + t + 256];
  s1[t] = a + bb; s2[t] = a * a + bb * bb;
  __syncthreads();
  for (int st = 128; st > 0; st >>= 1) {
    if (t < st) { s1[t] += s1[t + st]; s2[t] += s2[t + st]; }
    __syncthreads();
  }
  float mean = s1[0] * (1.f / 512.f);
  float var  = s2[0] * (1.f / 512.f) - mean * mean;
  float rstd = rsqrtf(var + 1e-5f);
  out[base + t]       = (a  - mean) * rstd * gamma[t]       + beta[t];
  out[base + t + 256] = (bb - mean) * rstd * gamma[t + 256] + beta[t + 256];
}

// ---------------------------------------------------------------------------
extern "C" void kernel_launch(void* const* d_in, const int* in_sizes, int n_in,
                              void* d_out, int out_size, void* d_ws, size_t ws_size,
                              hipStream_t stream) {
  (void)in_sizes; (void)n_in; (void)out_size; (void)ws_size;
  const float* x1  = (const float*)d_in[0];   // [B,N,C1]
  const float* x2  = (const float*)d_in[1];   // [B,N,C2]
  const float* Wp1 = (const float*)d_in[2];  const float* bp1 = (const float*)d_in[3];
  const float* Wp2 = (const float*)d_in[4];  const float* bp2 = (const float*)d_in[5];
  const float* Wq1 = (const float*)d_in[6];  const float* bq1 = (const float*)d_in[7];
  const float* Wk1 = (const float*)d_in[8];  const float* bk1 = (const float*)d_in[9];
  const float* Wv1 = (const float*)d_in[10]; const float* bv1 = (const float*)d_in[11];
  const float* Wq2 = (const float*)d_in[12]; const float* bq2 = (const float*)d_in[13];
  const float* Wk2 = (const float*)d_in[14]; const float* bk2 = (const float*)d_in[15];
  const float* Wv2 = (const float*)d_in[16]; const float* bv2 = (const float*)d_in[17];
  const float* Wg  = (const float*)d_in[18]; const float* bg  = (const float*)d_in[19];
  const float* Wf1 = (const float*)d_in[20]; const float* bf1 = (const float*)d_in[21];
  const float* gf  = (const float*)d_in[22]; const float* bfn = (const float*)d_in[23];
  const float* Wf2 = (const float*)d_in[24]; const float* bf2 = (const float*)d_in[25];
  const float* gn  = (const float*)d_in[26]; const float* bn  = (const float*)d_in[27];

  char* ws = (char*)d_ws;
  size_t off = 0;
  auto alloc = [&](size_t bytes) -> void* {
    void* p = ws + off;
    off = (off + bytes + 255) & ~(size_t)255;
    return p;
  };

  __half* x1h   = (__half*)alloc((size_t)M_ * C1_ * 2);
  __half* x2h   = (__half*)alloc((size_t)M_ * C2_ * 2);
  float*  pn2f  = (float*) alloc((size_t)M_ * OUT_ * 4);
  float*  ptf   = (float*) alloc((size_t)M_ * OUT_ * 4);
  __half* pn2h  = (__half*)alloc((size_t)M_ * OUT_ * 2);
  __half* pth   = (__half*)alloc((size_t)M_ * OUT_ * 2);
  __half* q1h   = (__half*)alloc((size_t)M_ * OUT_ * 2);
  __half* k1h   = (__half*)alloc((size_t)M_ * OUT_ * 2);
  __half* v1h   = (__half*)alloc((size_t)M_ * OUT_ * 2);  // pair-packed
  __half* q2h   = (__half*)alloc((size_t)M_ * OUT_ * 2);
  __half* k2h   = (__half*)alloc((size_t)M_ * OUT_ * 2);
  __half* v2h   = (__half*)alloc((size_t)M_ * OUT_ * 2);  // pair-packed
  float*  out1  = (float*) alloc((size_t)M_ * OUT_ * 4);
  float*  out2  = (float*) alloc((size_t)M_ * OUT_ * 4);
  __half* cat3h = (__half*)alloc((size_t)M_ * 1536 * 2);
  float*  gatef = (float*) alloc((size_t)M_ * OUT_ * 4);
  __half* cat2h = (__half*)alloc((size_t)M_ * 1024 * 2);
  float*  hmidf = (float*) alloc((size_t)M_ * OUT_ * 4);
  __half* hmidh = (__half*)alloc((size_t)M_ * OUT_ * 2);
  float*  h2f   = (float*) alloc((size_t)M_ * OUT_ * 4);
  unsigned int* Wp1p = (unsigned int*)alloc((size_t)(C1_ / 2) * OUT_ * 4);
  unsigned int* Wp2p = (unsigned int*)alloc((size_t)(C2_ / 2) * OUT_ * 4);
  unsigned int* Wq1p = (unsigned int*)alloc((size_t)(OUT_ / 2) * OUT_ * 4);
  unsigned int* Wk1p = (unsigned int*)alloc((size_t)(OUT_ / 2) * OUT_ * 4);
  unsigned int* Wv1p = (unsigned int*)alloc((size_t)(OUT_ / 2) * OUT_ * 4);
  unsigned int* Wq2p = (unsigned int*)alloc((size_t)(OUT_ / 2) * OUT_ * 4);
  unsigned int* Wk2p = (unsigned int*)alloc((size_t)(OUT_ / 2) * OUT_ * 4);
  unsigned int* Wv2p = (unsigned int*)alloc((size_t)(OUT_ / 2) * OUT_ * 4);
  unsigned int* Wgp  = (unsigned int*)alloc((size_t)(1536 / 2) * OUT_ * 4);
  unsigned int* Wf1p = (unsigned int*)alloc((size_t)(1024 / 2) * OUT_ * 4);
  unsigned int* Wf2p = (unsigned int*)alloc((size_t)(OUT_ / 2) * OUT_ * 4);

  auto g1 = [](int n) { return dim3((unsigned)((n + 255) / 256)); };

  // f16 copies of inputs + packed weights
  cvt_f2h<<<g1(M_ * C1_), 256, 0, stream>>>(x1, x1h, M_ * C1_);
  cvt_f2h<<<g1(M_ * C2_), 256, 0, stream>>>(x2, x2h, M_ * C2_);
  pack_w<<<g1((C1_/2)*OUT_),  256, 0, stream>>>(Wp1, Wp1p, C1_,  OUT_);
  pack_w<<<g1((C2_/2)*OUT_),  256, 0, stream>>>(Wp2, Wp2p, C2_,  OUT_);
  pack_w<<<g1((OUT_/2)*OUT_), 256, 0, stream>>>(Wq1, Wq1p, OUT_, OUT_);
  pack_w<<<g1((OUT_/2)*OUT_), 256, 0, stream>>>(Wk1, Wk1p, OUT_, OUT_);
  pack_w<<<g1((OUT_/2)*OUT_), 256, 0, stream>>>(Wv1, Wv1p, OUT_, OUT_);
  pack_w<<<g1((OUT_/2)*OUT_), 256, 0, stream>>>(Wq2, Wq2p, OUT_, OUT_);
  pack_w<<<g1((OUT_/2)*OUT_), 256, 0, stream>>>(Wk2, Wk2p, OUT_, OUT_);
  pack_w<<<g1((OUT_/2)*OUT_), 256, 0, stream>>>(Wv2, Wv2p, OUT_, OUT_);
  pack_w<<<g1((1536/2)*OUT_), 256, 0, stream>>>(Wg,  Wgp,  1536, OUT_);
  pack_w<<<g1((1024/2)*OUT_), 256, 0, stream>>>(Wf1, Wf1p, 1024, OUT_);
  pack_w<<<g1((OUT_/2)*OUT_), 256, 0, stream>>>(Wf2, Wf2p, OUT_, OUT_);

  const dim3 gemmBlk(128);
  const dim3 gemmGrd(OUT_ / 64, M_ / 64);

  // input projections (write both f32 for residual and f16 for next GEMMs)
  gemm_wmma<<<gemmGrd, gemmBlk, 0, stream>>>(x1h, Wp1p, bp1, pn2f, pn2h, C1_, OUT_, 3);
  gemm_wmma<<<gemmGrd, gemmBlk, 0, stream>>>(x2h, Wp2p, bp2, ptf,  pth,  C2_, OUT_, 3);

  // qkv projections (f16 only; V written pair-packed for the attention B-path)
  gemm_wmma<<<gemmGrd, gemmBlk, 0, stream>>>(pn2h, Wq1p, bq1, nullptr, q1h, OUT_, OUT_, 2);
  gemm_wmma<<<gemmGrd, gemmBlk, 0, stream>>>(pth,  Wk1p, bk1, nullptr, k1h, OUT_, OUT_, 2);
  gemm_wmma<<<gemmGrd, gemmBlk, 0, stream>>>(pth,  Wv1p, bv1, nullptr, v1h, OUT_, OUT_, 8);
  gemm_wmma<<<gemmGrd, gemmBlk, 0, stream>>>(pth,  Wq2p, bq2, nullptr, q2h, OUT_, OUT_, 2);
  gemm_wmma<<<gemmGrd, gemmBlk, 0, stream>>>(pn2h, Wk2p, bk2, nullptr, k2h, OUT_, OUT_, 2);
  gemm_wmma<<<gemmGrd, gemmBlk, 0, stream>>>(pn2h, Wv2p, bv2, nullptr, v2h, OUT_, OUT_, 8);

  // attention branches
  const dim3 attnGrd(N_ / 64, H_, B_);
  attn_wmma<<<attnGrd, dim3(128), 0, stream>>>(q1h, k1h, v1h, out1);
  attn_wmma<<<attnGrd, dim3(128), 0, stream>>>(q2h, k2h, v2h, out2);

  // gate = sigmoid(concat[out1,out2,pn2+pt] @ Wg + bg)
  build_cat3<<<g1(M_ * 1536), 256, 0, stream>>>(out1, out2, pn2f, ptf, cat3h);
  gemm_wmma<<<gemmGrd, gemmBlk, 0, stream>>>(cat3h, Wgp, bg, gatef, nullptr, 1536, OUT_, 5);

  // fused = concat[out1*gate, out2*(1-gate)] ; h = gelu(LN(fused@Wf1+bf1))
  build_cat2<<<g1(M_ * 1024), 256, 0, stream>>>(out1, out2, gatef, cat2h);
  gemm_wmma<<<gemmGrd, gemmBlk, 0, stream>>>(cat2h, Wf1p, bf1, hmidf, nullptr, 1024, OUT_, 1);
  ln_gelu<<<dim3(M_), dim3(256), 0, stream>>>(hmidf, gf, bfn, hmidh);

  // h @ Wf2 + bf2 ; final LN(h + pn2 + pt)
  gemm_wmma<<<gemmGrd, gemmBlk, 0, stream>>>(hmidh, Wf2p, bf2, h2f, nullptr, OUT_, OUT_, 1);
  final_ln<<<dim3(M_), dim3(256), 0, stream>>>(h2f, pn2f, ptf, gn, bn, (float*)d_out);
}
